// PhaseMultiHeadAttention_40372692582598
// MI455X (gfx1250) — matrix operand order repaired and checked by reading, not compiled, not run on hardware
//
#include <hip/hip_runtime.h>
#include <hip/hip_bf16.h>

// ---------------------------------------------------------------------------
// PhaseMultiHeadAttention for gfx1250 (CDNA5, wave32, WMMA 16x16x32 bf16)
//   1) f32 -> bf16 converts
//   2) QKV projection GEMMs (bf16 WMMA); Q/K in [B,H,N,dh], V in [B,H,dh,N]
//   3) fused flash-attention: K/V tiles double-buffered into LDS with
//      GLOBAL_LOAD_ASYNC_TO_LDS_B128 (ASYNCcnt), QK^T + skewed relative term
//      (Q*E_slice^T with LDS diagonal gather), online softmax, P*V, phase
//   4) output projection GEMM (bf16 WMMA, f32 out + bias)
// ---------------------------------------------------------------------------

typedef __bf16 bfx8  __attribute__((ext_vector_type(8)));
typedef __bf16 bfx16 __attribute__((ext_vector_type(16)));
typedef float  fx8   __attribute__((ext_vector_type(8)));

constexpr int Bb   = 8;
constexpr int Nn   = 1024;
constexpr int Dd   = 512;
constexpr int Hh   = 8;
constexpr int DH   = 64;
constexpr int Lrel = 2 * Nn - 1;      // 2047
constexpr float SCALE = 0.125f;       // 1/sqrt(dh)

#define DEVFN static __device__ __forceinline__

// 16 contiguous bf16 -> B-fragment (one lane's 16 K-slots)
DEVFN bfx16 ld16(const __bf16* p) {
  bfx8 lo = *reinterpret_cast<const bfx8*>(p);
  bfx8 hi = *reinterpret_cast<const bfx8*>(p + 8);
  return __builtin_shufflevector(lo, hi, 0, 1, 2, 3, 4, 5, 6, 7,
                                 8, 9, 10, 11, 12, 13, 14, 15);
}

// A-fragment lane load: elements [0..15] relative to p
DEVFN bfx16 ld_a(const __bf16* p) {
  bfx8 lo = *reinterpret_cast<const bfx8*>(p);
  bfx8 hi = *reinterpret_cast<const bfx8*>(p + 8);
  return __builtin_shufflevector(lo, hi, 0, 1, 2, 3, 4, 5, 6, 7,
                                 8, 9, 10, 11, 12, 13, 14, 15);
}

DEVFN fx8 wmma_bf16(bfx16 a, bfx16 b, fx8 c) {
  return __builtin_amdgcn_wmma_f32_16x16x32_bf16(false, a, false, b,
                                                 (short)0, c, false, false);
}

// Async copy 16B/lane global -> LDS (GV mode; LDS offset = low 32 bits of
// the generic shared address per ISA 10.2).
DEVFN void async_ld16B(const void* gaddr, void* lds_generic) {
  unsigned lds_off = (unsigned)(uintptr_t)lds_generic;
  unsigned long long ga = (unsigned long long)(uintptr_t)gaddr;
  asm volatile("global_load_async_to_lds_b128 %0, %1, off"
               :
               : "v"(lds_off), "v"(ga)
               : "memory");
}
DEVFN void wait_async0() {
  asm volatile("s_wait_asynccnt 0x0" ::: "memory");
}

// ---------------------------------------------------------------------------
__global__ void f32_to_bf16(const float* __restrict__ src,
                            __bf16* __restrict__ dst, int n) {
  int i = blockIdx.x * blockDim.x + threadIdx.x;
  int stride = gridDim.x * blockDim.x;
  for (; i < n; i += stride) dst[i] = (__bf16)src[i];
}

// ---------------------------------------------------------------------------
// C[16x64 per wave] = A[M,512] * W[512,512]^T   (both row-major, K contiguous)
// MODE 0: bf16 head layout [B,H,N,dh] (Q,K) | MODE 1: bf16 [B,H,dh,N] (V)
// MODE 2: f32 [M,512] + bias (output projection)
template <int MODE>
__global__ __launch_bounds__(256) void gemm_proj(
    const __bf16* __restrict__ A, const __bf16* __restrict__ W,
    __bf16* __restrict__ outb, float* __restrict__ outf,
    const float* __restrict__ bias) {
  const int lane = threadIdx.x & 31;
  const int wave = threadIdx.x >> 5;
  const int g  = lane >> 4;
  const int ln = lane & 15;
  const int row0 = blockIdx.x * 128 + wave * 16;
  const int col0 = blockIdx.y * 64;

  fx8 acc[4] = {};
  const __bf16* ap = A + (size_t)(row0 + ln) * Dd;

  for (int k0 = 0; k0 < Dd; k0 += 32) {
    bfx16 af = ld_a(ap + k0 + g * 16);
#pragma unroll
    for (int j = 0; j < 4; ++j) {
      bfx16 bf = ld16(W + (size_t)(col0 + j * 16 + ln) * Dd + k0 + g * 16);
      acc[j] = wmma_bf16(af, bf, acc[j]);
    }
  }

#pragma unroll
  for (int j = 0; j < 4; ++j) {
#pragma unroll
    for (int v = 0; v < 8; ++v) {
      const int row = row0 + v + 8 * g;
      const int col = col0 + j * 16 + ln;
      const float val = acc[j][v];
      if (MODE == 2) {
        outf[(size_t)row * Dd + col] = val + bias[col];
      } else {
        const int b = row >> 10, n = row & 1023;
        const int h = col >> 6, d = col & 63;
        size_t idx;
        if (MODE == 0)
          idx = (((size_t)(b * Hh + h) * Nn) + n) * DH + d;
        else
          idx = (((size_t)(b * Hh + h) * DH) + d) * Nn + n;
        outb[idx] = (__bf16)val;
      }
    }
  }
}

// ---------------------------------------------------------------------------
// Fused attention. Block = 8 waves = 128 n-rows of one (b,h); m streamed in
// steps of 32 with K/V tiles async-staged into double-buffered LDS.
__global__ __launch_bounds__(256) void attn_kernel(
    const __bf16* __restrict__ Q, const __bf16* __restrict__ K,
    const __bf16* __restrict__ Vt, const __bf16* __restrict__ E,
    const float* __restrict__ phase, __bf16* __restrict__ O) {
  __shared__ __align__(16) __bf16 ldsK[2][32 * DH];   // 2 x 4KB (32 m x 64 d)
  __shared__ __align__(16) __bf16 ldsV[2][DH * 32];   // 2 x 4KB (64 d x 32 m)
  __shared__ __align__(16) float  ldsT[8][16 * 48];   // Q*E tiles, per-wave
  __shared__ __align__(16) __bf16 ldsP[8][16 * 32];   // probs, per-wave

  const int tid  = threadIdx.x;
  const int lane = tid & 31;
  const int wave = tid >> 5;
  const int g  = lane >> 4;
  const int ln = lane & 15;
  const int bh = blockIdx.y;
  const int n0 = blockIdx.x * 128 + wave * 16;

  const __bf16* Qh = Q + (size_t)bh * Nn * DH;
  const __bf16* Kh = K + (size_t)bh * Nn * DH;
  const __bf16* Vh = Vt + (size_t)bh * DH * Nn;

  // cooperative staging slices: one 16B async transfer per lane per tile
  const int kr = tid >> 3, kc = tid & 7;  // K: row 0..31, 8B-elem chunk 0..7
  const int vd = tid >> 2, vc = tid & 3;  // V: row 0..63, chunk 0..3

  // Q A-fragments (reused for all m-steps and for the E-slice products)
  bfx16 qa[2];
#pragma unroll
  for (int kd = 0; kd < 2; ++kd)
    qa[kd] = ld_a(Qh + (size_t)(n0 + ln) * DH + kd * 32 + g * 16);

  fx8 acc[4] = {};
  float row_m[8], row_l[8];
#pragma unroll
  for (int v = 0; v < 8; ++v) { row_m[v] = -3.0e38f; row_l[v] = 0.0f; }

  // stage first K/V tiles
  async_ld16B(Kh + (size_t)kr * DH + kc * 8, &ldsK[0][kr * DH + kc * 8]);
  async_ld16B(Vh + (size_t)vd * Nn + vc * 8, &ldsV[0][vd * 32 + vc * 8]);

  for (int m0 = 0; m0 < Nn; m0 += 32) {
    const int buf = (m0 >> 5) & 1;
    wait_async0();      // my slice of the current tiles has landed
    __syncthreads();    // everyone's slice has landed
    if (m0 + 32 < Nn) { // refill the other buffer while we compute
      const int ms = m0 + 32;
      async_ld16B(Kh + (size_t)(ms + kr) * DH + kc * 8,
                  &ldsK[buf ^ 1][kr * DH + kc * 8]);
      async_ld16B(Vh + (size_t)vd * Nn + ms + vc * 8,
                  &ldsV[buf ^ 1][vd * 32 + vc * 8]);
    }

    // ---- S = Q K^T from LDS-staged K tile
    fx8 s[2];
#pragma unroll
    for (int t = 0; t < 2; ++t) {
      fx8 c = {};
#pragma unroll
      for (int kd = 0; kd < 2; ++kd) {
        bfx16 bf = ld16(&ldsK[buf][(t * 16 + ln) * DH + kd * 32 + g * 16]);
        c = wmma_bf16(qa[kd], bf, c);
      }
      s[t] = c;
    }

    // ---- relative term: T[n][j] = Q[n] . E[lbase + j], j in [0,48)
    const int lbase = m0 - n0 + 1008;  // >= 0 always
    __builtin_prefetch((const void*)(E + (size_t)(lbase < Lrel - 48 ? lbase + 48
                                                                    : lbase) *
                                             DH),
                       0, 0);
#pragma unroll
    for (int c3 = 0; c3 < 3; ++c3) {
      int erow = lbase + c3 * 16 + ln;
      erow = erow > (Lrel - 1) ? (Lrel - 1) : erow;  // clamp padding column
      fx8 tacc = {};
#pragma unroll
      for (int kd = 0; kd < 2; ++kd) {
        bfx16 bf = ld16(E + (size_t)erow * DH + kd * 32 + g * 16);
        tacc = wmma_bf16(qa[kd], bf, tacc);
      }
#pragma unroll
      for (int v = 0; v < 8; ++v)
        ldsT[wave][(v + 8 * g) * 48 + c3 * 16 + ln] = tacc[v];
    }

    // ---- skew gather + scale + online softmax
#pragma unroll
    for (int v = 0; v < 8; ++v) {
      const int nl = v + 8 * g;
#pragma unroll
      for (int t = 0; t < 2; ++t) {
        const int j = t * 16 + ln - nl + 15;
        s[t][v] = (s[t][v] + ldsT[wave][nl * 48 + j]) * SCALE;
      }
      float mx = fmaxf(s[0][v], s[1][v]);
#pragma unroll
      for (int off = 8; off > 0; off >>= 1)
        mx = fmaxf(mx, __shfl_xor(mx, off, 32));
      const float mnew = fmaxf(row_m[v], mx);
      const float corr = __expf(row_m[v] - mnew);
      row_m[v] = mnew;
      const float p0 = __expf(s[0][v] - mnew);
      const float p1 = __expf(s[1][v] - mnew);
      float ps = p0 + p1;
#pragma unroll
      for (int off = 8; off > 0; off >>= 1)
        ps += __shfl_xor(ps, off, 32);
      row_l[v] = row_l[v] * corr + ps;
#pragma unroll
      for (int j2 = 0; j2 < 4; ++j2) acc[j2][v] *= corr;
      ldsP[wave][nl * 32 + ln]      = (__bf16)p0;
      ldsP[wave][nl * 32 + 16 + ln] = (__bf16)p1;
    }

    // ---- redistribute P (C-layout) -> A-fragment layout via LDS
    bfx8 plo = *reinterpret_cast<const bfx8*>(&ldsP[wave][ln * 32 + g * 16]);
    bfx8 phi = *reinterpret_cast<const bfx8*>(&ldsP[wave][ln * 32 + g * 16 + 8]);
    bfx16 pa = __builtin_shufflevector(plo, phi, 0, 1, 2, 3, 4, 5, 6, 7,
                                       8, 9, 10, 11, 12, 13, 14, 15);

    // ---- O += P * V from LDS-staged V^T tile
#pragma unroll
    for (int j2 = 0; j2 < 4; ++j2) {
      bfx16 bf = ld16(&ldsV[buf][(j2 * 16 + ln) * 32 + g * 16]);
      acc[j2] = wmma_bf16(pa, bf, acc[j2]);
    }

    __syncthreads();  // done reading `buf` before it is refilled next iter
  }

  // ---- epilogue: normalize, phase-modulate, store [B, N, H*dh] bf16
  const int b = bh >> 3, h = bh & 7;
#pragma unroll
  for (int v = 0; v < 8; ++v) {
    const int nl = v + 8 * g;
    const int n = n0 + nl;
    const float w = phase[(size_t)bh * Nn + n] / row_l[v];
#pragma unroll
    for (int j2 = 0; j2 < 4; ++j2) {
      const int d = j2 * 16 + ln;
      O[((size_t)(b * Nn + n)) * (Hh * DH) + h * DH + d] =
          (__bf16)(acc[j2][v] * w);
    }
  }
}

// ---------------------------------------------------------------------------
extern "C" void kernel_launch(void* const* d_in, const int* in_sizes, int n_in,
                              void* d_out, int out_size, void* d_ws,
                              size_t ws_size, hipStream_t stream) {
  (void)in_sizes; (void)n_in; (void)out_size; (void)ws_size;
  const float* x     = (const float*)d_in[0];
  const float* phase = (const float*)d_in[1];
  const float* E_rel = (const float*)d_in[2];
  const float* W_q   = (const float*)d_in[3];
  const float* W_k   = (const float*)d_in[4];
  const float* W_v   = (const float*)d_in[5];
  const float* W_o   = (const float*)d_in[6];
  const float* b_o   = (const float*)d_in[7];
  float* out = (float*)d_out;

  char* ws = (char*)d_ws;
  size_t off = 0;
  auto alloc = [&](size_t bytes) -> char* {
    char* p = ws + off;
    off += (bytes + 255) & ~(size_t)255;
    return p;
  };
  const size_t nX = (size_t)Bb * Nn * Dd;
  const size_t nW = (size_t)Dd * Dd;
  const size_t nE = (size_t)Lrel * DH;
  const size_t nQ = (size_t)Bb * Hh * Nn * DH;

  __bf16* xb  = (__bf16*)alloc(nX * 2);
  __bf16* wqb = (__bf16*)alloc(nW * 2);
  __bf16* wkb = (__bf16*)alloc(nW * 2);
  __bf16* wvb = (__bf16*)alloc(nW * 2);
  __bf16* wob = (__bf16*)alloc(nW * 2);
  __bf16* eb  = (__bf16*)alloc(nE * 2);
  __bf16* Qb  = (__bf16*)alloc(nQ * 2);
  __bf16* Kb  = (__bf16*)alloc(nQ * 2);
  __bf16* Vtb = (__bf16*)alloc(nQ * 2);
  __bf16* Ob  = (__bf16*)alloc(nX * 2);

  f32_to_bf16<<<1024, 256, 0, stream>>>(x, xb, (int)nX);
  f32_to_bf16<<<256, 256, 0, stream>>>(W_q, wqb, (int)nW);
  f32_to_bf16<<<256, 256, 0, stream>>>(W_k, wkb, (int)nW);
  f32_to_bf16<<<256, 256, 0, stream>>>(W_v, wvb, (int)nW);
  f32_to_bf16<<<256, 256, 0, stream>>>(W_o, wob, (int)nW);
  f32_to_bf16<<<256, 256, 0, stream>>>(E_rel, eb, (int)nE);

  dim3 gg(Bb * Nn / 128, Dd / 64), bb(256);
  gemm_proj<0><<<gg, bb, 0, stream>>>(xb, wqb, Qb, nullptr, nullptr);
  gemm_proj<0><<<gg, bb, 0, stream>>>(xb, wkb, Kb, nullptr, nullptr);
  gemm_proj<1><<<gg, bb, 0, stream>>>(xb, wvb, Vtb, nullptr, nullptr);

  attn_kernel<<<dim3(Nn / 128, Bb * Hh), 256, 0, stream>>>(Qb, Kb, Vtb, eb,
                                                           phase, Ob);

  gemm_proj<2><<<gg, bb, 0, stream>>>(Ob, wob, nullptr, out, b_o);
}
